// GNN_28217935135266
// MI455X (gfx1250) — compile-verified, matrix operand order checked by reading, not run on hardware
//
#include <hip/hip_runtime.h>
#include <hip/hip_bf16.h>

// ---------------------------------------------------------------------------
// GNN forward for MI455X (gfx1250, wave32, WMMA 16x16x32 f16 -> f32 acc)
// ---------------------------------------------------------------------------
#define N_NODES 50000
#define N_EDGES 800000
#define N_GRAPH 512
#define IN_C    16
#define H       128
#define EC      16
#define NLAYERS 4

typedef __attribute__((ext_vector_type(16))) _Float16 v16h;
typedef __attribute__((ext_vector_type(8)))  float    v8f;

#define WMMA_F16(a, b, c) \
  __builtin_amdgcn_wmma_f32_16x16x32_f16(false, (a), false, (b), (short)0, (c), false, false)

__device__ __forceinline__ float silu_f(float x) {
  return x / (1.0f + __expf(-x));
}

__device__ __forceinline__ v16h loadh(const _Float16* p) {
  return *(const v16h*)p;  // 32B contiguous -> 2x b128 loads
}

__device__ __forceinline__ v16h cvt16(const float* p) {
  v16h r;
#pragma unroll
  for (int i = 0; i < 16; ++i) r[i] = (_Float16)p[i];
  return r;
}

// ---------------------------------------------------------------------------
// Weight prep: w[K][Nout] (f32) -> wt[Nout][Kpad] (f16, transposed, 0-padded)
// ---------------------------------------------------------------------------
__global__ void k_prep_wt(const float* __restrict__ w, _Float16* __restrict__ wt,
                          int K, int Nout, int Kpad) {
  int i = blockIdx.x * 256 + threadIdx.x;
  if (i >= Nout * Kpad) return;
  int n = i / Kpad, k = i - n * Kpad;
  wt[i] = (k < K) ? (_Float16)w[(size_t)k * Nout + n] : (_Float16)0.0f;
}

__global__ void k_cvt_f16(const float* __restrict__ src, _Float16* __restrict__ dst, int n) {
  int i = blockIdx.x * 256 + threadIdx.x;
  if (i < n) dst[i] = (_Float16)src[i];
}

__global__ void k_zero(float* __restrict__ p, int n) {
  int i = blockIdx.x * 256 + threadIdx.x;
  if (i < n) p[i] = 0.0f;
}

// ---------------------------------------------------------------------------
// Embed: h = silu(x[N,16] @ W1[16,128] + b1) @ W2[128,128] + b2   (h stored f16)
// 8 waves/block, 16 nodes per wave.
// ---------------------------------------------------------------------------
__global__ void k_embed(const float* __restrict__ xn,
                        const _Float16* __restrict__ w1t,  // [128][32]
                        const float* __restrict__ b1,
                        const _Float16* __restrict__ w2t,  // [128][128]
                        const float* __restrict__ b2,
                        _Float16* __restrict__ hf, int Ncnt) {
  __shared__ alignas(32) _Float16 lds[8][16 * 128];
  const int wave = threadIdx.x >> 5, lane = threadIdx.x & 31;
  const int lrow = lane & 15, hi = lane >> 4, kh = hi * 16;
  const int m0 = (blockIdx.x * 8 + wave) * 16;
  const int row = m0 + lrow;
  const int rowc = row < Ncnt ? row : Ncnt - 1;

  v16h a = {};
  if (hi == 0) a = cvt16(xn + (size_t)rowc * IN_C);  // K pad 16 -> 32

  _Float16* t = &lds[wave][0];
#pragma unroll
  for (int nt = 0; nt < 8; ++nt) {
    v8f c = {};
    c = WMMA_F16(a, loadh(w1t + (nt * 16 + lrow) * 32 + kh), c);
    const float bias = b1[nt * 16 + lrow];
#pragma unroll
    for (int r = 0; r < 8; ++r)
      t[(r + hi * 8) * 128 + nt * 16 + lrow] = (_Float16)silu_f(c[r] + bias);
  }

  v16h a2[4];
#pragma unroll
  for (int kt = 0; kt < 4; ++kt) a2[kt] = loadh(t + lrow * 128 + kt * 32 + kh);

#pragma unroll
  for (int nt = 0; nt < 8; ++nt) {
    v8f c = {};
#pragma unroll
    for (int kt = 0; kt < 4; ++kt)
      c = WMMA_F16(a2[kt], loadh(w2t + (nt * 16 + lrow) * 128 + kt * 32 + kh), c);
    const float bias = b2[nt * 16 + lrow];
#pragma unroll
    for (int r = 0; r < 8; ++r) {
      const int m = m0 + r + hi * 8;
      if (m < Ncnt) hf[(size_t)m * 128 + nt * 16 + lrow] = (_Float16)(c[r] + bias);
    }
  }
}

// ---------------------------------------------------------------------------
// Edge MLP + scatter: msg = silu(silu([h_s|h_r|xe] @ W1 + b1) @ W2 + b2)
// aggr[receiver] += msg   (global f32 atomics)
// K = 272 padded to 288 (9 k-tiles). 8 waves/block, 16 edges/wave. E%128==0.
// ---------------------------------------------------------------------------
__global__ void k_edge(const _Float16* __restrict__ hf,
                       const _Float16* __restrict__ xe,   // [E][16] f16
                       const int* __restrict__ ei,        // [2][E]
                       const _Float16* __restrict__ w1t,  // [128][288]
                       const float* __restrict__ b1,
                       const _Float16* __restrict__ w2t,  // [128][128]
                       const float* __restrict__ b2,
                       float* __restrict__ aggr, int Ecnt) {
  __shared__ alignas(32) _Float16 lds[8][16 * 128];
  __shared__ int rcvs[8][16];
  const int wave = threadIdx.x >> 5, lane = threadIdx.x & 31;
  const int lrow = lane & 15, hi = lane >> 4, kh = hi * 16;
  const int m0 = (blockIdx.x * 8 + wave) * 16;
  const int row = m0 + lrow;

  const int snd = ei[row];
  const int rcv = ei[Ecnt + row];
  if (hi == 0) rcvs[wave][lrow] = rcv;

  // A fragments: 4 sender tiles, 4 receiver tiles, 1 edge-feature tile.
  v16h a[9];
  const _Float16* hs = hf + (size_t)snd * 128;
  const _Float16* hr = hf + (size_t)rcv * 128;
#pragma unroll
  for (int kt = 0; kt < 4; ++kt) a[kt] = loadh(hs + kt * 32 + kh);
#pragma unroll
  for (int kt = 0; kt < 4; ++kt) a[4 + kt] = loadh(hr + kt * 32 + kh);
  {
    v16h z = {};
    a[8] = (hi == 0) ? loadh(xe + (size_t)row * EC) : z;  // k 256..271, rest zero
  }

  _Float16* t = &lds[wave][0];
#pragma unroll
  for (int nt = 0; nt < 8; ++nt) {
    v8f c = {};
    const _Float16* wb = w1t + (nt * 16 + lrow) * 288 + kh;
#pragma unroll
    for (int kt = 0; kt < 9; ++kt) c = WMMA_F16(a[kt], loadh(wb + kt * 32), c);
    const float bias = b1[nt * 16 + lrow];
#pragma unroll
    for (int r = 0; r < 8; ++r)
      t[(r + hi * 8) * 128 + nt * 16 + lrow] = (_Float16)silu_f(c[r] + bias);
  }

  v16h a2[4];
#pragma unroll
  for (int kt = 0; kt < 4; ++kt) a2[kt] = loadh(t + lrow * 128 + kt * 32 + kh);

#pragma unroll
  for (int nt = 0; nt < 8; ++nt) {
    v8f c = {};
#pragma unroll
    for (int kt = 0; kt < 4; ++kt)
      c = WMMA_F16(a2[kt], loadh(w2t + (nt * 16 + lrow) * 128 + kt * 32 + kh), c);
    const float bias = b2[nt * 16 + lrow];
#pragma unroll
    for (int r = 0; r < 8; ++r) {
      const float x = silu_f(c[r] + bias);
      const int rv = rcvs[wave][r + hi * 8];
      atomicAdd(&aggr[(size_t)rv * 128 + nt * 16 + lrow], x);
    }
  }
}

// ---------------------------------------------------------------------------
// Node MLP: h = silu([h|aggr] @ W1[256,128] + b1) @ W2 + b2   (in-place on hf)
// ---------------------------------------------------------------------------
__global__ void k_node(_Float16* __restrict__ hf,
                       const float* __restrict__ aggr,
                       const _Float16* __restrict__ w1t,  // [128][256]
                       const float* __restrict__ b1,
                       const _Float16* __restrict__ w2t,  // [128][128]
                       const float* __restrict__ b2, int Ncnt) {
  __shared__ alignas(32) _Float16 lds[8][16 * 128];
  const int wave = threadIdx.x >> 5, lane = threadIdx.x & 31;
  const int lrow = lane & 15, hi = lane >> 4, kh = hi * 16;
  const int m0 = (blockIdx.x * 8 + wave) * 16;
  const int row = m0 + lrow;
  const int rowc = row < Ncnt ? row : Ncnt - 1;

  v16h a[8];
  const _Float16* hr = hf + (size_t)rowc * 128;
  const float* ar = aggr + (size_t)rowc * 128;
#pragma unroll
  for (int kt = 0; kt < 4; ++kt) a[kt] = loadh(hr + kt * 32 + kh);
#pragma unroll
  for (int kt = 0; kt < 4; ++kt) a[4 + kt] = cvt16(ar + kt * 32 + kh);

  _Float16* t = &lds[wave][0];
#pragma unroll
  for (int nt = 0; nt < 8; ++nt) {
    v8f c = {};
    const _Float16* wb = w1t + (nt * 16 + lrow) * 256 + kh;
#pragma unroll
    for (int kt = 0; kt < 8; ++kt) c = WMMA_F16(a[kt], loadh(wb + kt * 32), c);
    const float bias = b1[nt * 16 + lrow];
#pragma unroll
    for (int r = 0; r < 8; ++r)
      t[(r + hi * 8) * 128 + nt * 16 + lrow] = (_Float16)silu_f(c[r] + bias);
  }

  v16h a2[4];
#pragma unroll
  for (int kt = 0; kt < 4; ++kt) a2[kt] = loadh(t + lrow * 128 + kt * 32 + kh);

#pragma unroll
  for (int nt = 0; nt < 8; ++nt) {
    v8f c = {};
#pragma unroll
    for (int kt = 0; kt < 4; ++kt)
      c = WMMA_F16(a2[kt], loadh(w2t + (nt * 16 + lrow) * 128 + kt * 32 + kh), c);
    const float bias = b2[nt * 16 + lrow];
#pragma unroll
    for (int r = 0; r < 8; ++r) {
      const int m = m0 + r + hi * 8;
      if (m < Ncnt) hf[(size_t)m * 128 + nt * 16 + lrow] = (_Float16)(c[r] + bias);
    }
  }
}

// ---------------------------------------------------------------------------
// Pre-readout + pooling: y = silu(h @ W1 + b1) @ W2 + b2 ; pooled[batch] += y
// ---------------------------------------------------------------------------
__global__ void k_prereadout(const _Float16* __restrict__ hf,
                             const int* __restrict__ batch,
                             const _Float16* __restrict__ w1t,
                             const float* __restrict__ b1,
                             const _Float16* __restrict__ w2t,
                             const float* __restrict__ b2,
                             float* __restrict__ pooled, int Ncnt) {
  __shared__ alignas(32) _Float16 lds[8][16 * 128];
  __shared__ int bids[8][16];
  const int wave = threadIdx.x >> 5, lane = threadIdx.x & 31;
  const int lrow = lane & 15, hi = lane >> 4, kh = hi * 16;
  const int m0 = (blockIdx.x * 8 + wave) * 16;
  const int row = m0 + lrow;
  const int rowc = row < Ncnt ? row : Ncnt - 1;
  if (hi == 0) bids[wave][lrow] = batch[rowc];

  v16h a[4];
  const _Float16* hr = hf + (size_t)rowc * 128;
#pragma unroll
  for (int kt = 0; kt < 4; ++kt) a[kt] = loadh(hr + kt * 32 + kh);

  _Float16* t = &lds[wave][0];
#pragma unroll
  for (int nt = 0; nt < 8; ++nt) {
    v8f c = {};
#pragma unroll
    for (int kt = 0; kt < 4; ++kt)
      c = WMMA_F16(a[kt], loadh(w1t + (nt * 16 + lrow) * 128 + kt * 32 + kh), c);
    const float bias = b1[nt * 16 + lrow];
#pragma unroll
    for (int r = 0; r < 8; ++r)
      t[(r + hi * 8) * 128 + nt * 16 + lrow] = (_Float16)silu_f(c[r] + bias);
  }

  v16h a2[4];
#pragma unroll
  for (int kt = 0; kt < 4; ++kt) a2[kt] = loadh(t + lrow * 128 + kt * 32 + kh);

#pragma unroll
  for (int nt = 0; nt < 8; ++nt) {
    v8f c = {};
#pragma unroll
    for (int kt = 0; kt < 4; ++kt)
      c = WMMA_F16(a2[kt], loadh(w2t + (nt * 16 + lrow) * 128 + kt * 32 + kh), c);
    const float bias = b2[nt * 16 + lrow];
#pragma unroll
    for (int r = 0; r < 8; ++r) {
      const int m = m0 + r + hi * 8;
      if (m < Ncnt) {
        const int g = bids[wave][r + hi * 8];
        atomicAdd(&pooled[(size_t)g * 128 + nt * 16 + lrow], c[r] + bias);
      }
    }
  }
}

// ---------------------------------------------------------------------------
// Readout: out = silu(pooled @ W1 + b1) @ w2[128,1] + b2 ; 1 wave = 16 graphs.
// ---------------------------------------------------------------------------
__global__ void k_readout(const float* __restrict__ pooled,
                          const _Float16* __restrict__ w1t,
                          const float* __restrict__ b1,
                          const float* __restrict__ w2,  // [128]
                          const float* __restrict__ b2,  // [1]
                          float* __restrict__ out) {
  __shared__ alignas(32) float ts[16 * 128];
  const int lane = threadIdx.x & 31;
  const int lrow = lane & 15, hi = lane >> 4, kh = hi * 16;
  const int m0 = blockIdx.x * 16;

  v16h a[4];
  const float* pr = pooled + (size_t)(m0 + lrow) * 128;
#pragma unroll
  for (int kt = 0; kt < 4; ++kt) a[kt] = cvt16(pr + kt * 32 + kh);

#pragma unroll
  for (int nt = 0; nt < 8; ++nt) {
    v8f c = {};
#pragma unroll
    for (int kt = 0; kt < 4; ++kt)
      c = WMMA_F16(a[kt], loadh(w1t + (nt * 16 + lrow) * 128 + kt * 32 + kh), c);
    const float bias = b1[nt * 16 + lrow];
#pragma unroll
    for (int r = 0; r < 8; ++r)
      ts[(r + hi * 8) * 128 + nt * 16 + lrow] = silu_f(c[r] + bias);
  }

  if (lane < 16) {
    float s = b2[0];
#pragma unroll 8
    for (int n = 0; n < 128; ++n) s += ts[lane * 128 + n] * w2[n];
    out[m0 + lane] = s;
  }
}

// ---------------------------------------------------------------------------
extern "C" void kernel_launch(void* const* d_in, const int* in_sizes, int n_in,
                              void* d_out, int out_size, void* d_ws, size_t ws_size,
                              hipStream_t stream) {
  (void)in_sizes; (void)n_in; (void)out_size; (void)ws_size;

  const float* x_nodes = (const float*)d_in[0];
  const float* x_edges = (const float*)d_in[1];
  const int*   edge_ix = (const int*)d_in[2];
  const int*   batch   = (const int*)d_in[3];
  const float* emb_w1  = (const float*)d_in[4];
  const float* emb_b1  = (const float*)d_in[5];
  const float* emb_w2  = (const float*)d_in[6];
  const float* emb_b2  = (const float*)d_in[7];
  const float* ew1 = (const float*)d_in[8];
  const float* eb1 = (const float*)d_in[9];
  const float* ew2 = (const float*)d_in[10];
  const float* eb2 = (const float*)d_in[11];
  const float* nw1 = (const float*)d_in[12];
  const float* nb1 = (const float*)d_in[13];
  const float* nw2 = (const float*)d_in[14];
  const float* nb2 = (const float*)d_in[15];
  const float* pr_w1 = (const float*)d_in[16];
  const float* pr_b1 = (const float*)d_in[17];
  const float* pr_w2 = (const float*)d_in[18];
  const float* pr_b2 = (const float*)d_in[19];
  const float* ro_w1 = (const float*)d_in[20];
  const float* ro_b1 = (const float*)d_in[21];
  const float* ro_w2 = (const float*)d_in[22];
  const float* ro_b2 = (const float*)d_in[23];
  float* out = (float*)d_out;

  // ---- workspace carve (256B aligned) ----
  char* wp = (char*)d_ws;
  auto carve = [&](size_t bytes) -> void* {
    void* p = (void*)wp;
    wp += (bytes + 255) & ~(size_t)255;
    return p;
  };
  _Float16* hf     = (_Float16*)carve((size_t)N_NODES * 128 * 2);
  _Float16* xe16   = (_Float16*)carve((size_t)N_EDGES * EC * 2);
  float*    aggr   = (float*)   carve((size_t)N_NODES * 128 * 4);
  float*    pooled = (float*)   carve((size_t)N_GRAPH * 128 * 4);
  _Float16* embw1t = (_Float16*)carve(128 * 32 * 2);
  _Float16* embw2t = (_Float16*)carve(128 * 128 * 2);
  _Float16* ew1t   = (_Float16*)carve((size_t)NLAYERS * 128 * 288 * 2);
  _Float16* ew2t   = (_Float16*)carve((size_t)NLAYERS * 128 * 128 * 2);
  _Float16* nw1t   = (_Float16*)carve((size_t)NLAYERS * 128 * 256 * 2);
  _Float16* nw2t   = (_Float16*)carve((size_t)NLAYERS * 128 * 128 * 2);
  _Float16* prw1t  = (_Float16*)carve(128 * 128 * 2);
  _Float16* prw2t  = (_Float16*)carve(128 * 128 * 2);
  _Float16* row1t  = (_Float16*)carve(128 * 128 * 2);

  auto gP = [](int n) { return (n + 255) / 256; };

  // ---- weight prep ----
  k_prep_wt<<<gP(128 * 32), 256, 0, stream>>>(emb_w1, embw1t, 16, 128, 32);
  k_prep_wt<<<gP(128 * 128), 256, 0, stream>>>(emb_w2, embw2t, 128, 128, 128);
  for (int l = 0; l < NLAYERS; ++l) {
    k_prep_wt<<<gP(128 * 288), 256, 0, stream>>>(ew1 + (size_t)l * 272 * 128,
                                                 ew1t + (size_t)l * 128 * 288, 272, 128, 288);
    k_prep_wt<<<gP(128 * 128), 256, 0, stream>>>(ew2 + (size_t)l * 128 * 128,
                                                 ew2t + (size_t)l * 128 * 128, 128, 128, 128);
    k_prep_wt<<<gP(128 * 256), 256, 0, stream>>>(nw1 + (size_t)l * 256 * 128,
                                                 nw1t + (size_t)l * 128 * 256, 256, 128, 256);
    k_prep_wt<<<gP(128 * 128), 256, 0, stream>>>(nw2 + (size_t)l * 128 * 128,
                                                 nw2t + (size_t)l * 128 * 128, 128, 128, 128);
  }
  k_prep_wt<<<gP(128 * 128), 256, 0, stream>>>(pr_w1, prw1t, 128, 128, 128);
  k_prep_wt<<<gP(128 * 128), 256, 0, stream>>>(pr_w2, prw2t, 128, 128, 128);
  k_prep_wt<<<gP(128 * 128), 256, 0, stream>>>(ro_w1, row1t, 128, 128, 128);
  k_cvt_f16<<<gP(N_EDGES * EC), 256, 0, stream>>>(x_edges, xe16, N_EDGES * EC);

  // ---- embed ----
  const int nodeBlocks = ((N_NODES + 15) / 16 + 7) / 8;  // 391
  k_embed<<<nodeBlocks, 256, 0, stream>>>(x_nodes, embw1t, emb_b1, embw2t, emb_b2,
                                          hf, N_NODES);

  // ---- message-passing layers ----
  const int edgeBlocks = N_EDGES / 128;  // 6250 (E % 128 == 0)
  for (int l = 0; l < NLAYERS; ++l) {
    k_zero<<<gP(N_NODES * 128), 256, 0, stream>>>(aggr, N_NODES * 128);
    k_edge<<<edgeBlocks, 256, 0, stream>>>(hf, xe16, edge_ix,
                                           ew1t + (size_t)l * 128 * 288, eb1 + l * 128,
                                           ew2t + (size_t)l * 128 * 128, eb2 + l * 128,
                                           aggr, N_EDGES);
    k_node<<<nodeBlocks, 256, 0, stream>>>(hf, aggr,
                                           nw1t + (size_t)l * 128 * 256, nb1 + l * 128,
                                           nw2t + (size_t)l * 128 * 128, nb2 + l * 128,
                                           N_NODES);
  }

  // ---- pre-readout + pool + readout ----
  k_zero<<<gP(N_GRAPH * 128), 256, 0, stream>>>(pooled, N_GRAPH * 128);
  k_prereadout<<<nodeBlocks, 256, 0, stream>>>(hf, batch, prw1t, pr_b1, prw2t, pr_b2,
                                               pooled, N_NODES);
  k_readout<<<N_GRAPH / 16, 32, 0, stream>>>(pooled, row1t, ro_b1, ro_w2, ro_b2, out);
}
